// TorchFeedForwardPolicy_9534827397234
// MI455X (gfx1250) — matrix operand order repaired and checked by reading, not compiled, not run on hardware
//
#include <hip/hip_runtime.h>
#include <hip/hip_bf16.h>

// ---------------------------------------------------------------------------
// Sparsely-wired MLP policy, densified onto the CDNA5 WMMA path.
//
//   acts (B x 2048, bf16, LDS-resident per 32-batch tile, padded rows)
//   layer l: acts[:, K_l : K_l+320] = tanh(acts[:, :K_l] @ Wdense_l + b_l)
//            via v_wmma_f32_16x16x32_bf16, K_l = 128 + 320*l
//   output : 18-wide gather-dot (tiny; VALU)
//
// Register blocking: each wave owns a 32x32 C super-tile (2x2 WMMA tiles,
// 4 accumulators). Each A fragment (ds_load_b128 x2) feeds 2 WMMAs and each
// B fragment (global_load_b128 x2) feeds 2 WMMAs -> half the LDS + L2 bytes
// per v_wmma vs 1x1 tiling, which matters because 1 KB A + 1 KB B per WMMA
// would otherwise saturate the 256 B/cycle LDS banks long before the matrix
// pipes. 10 waves x (2 bt x 2 nt) covers the 2x20 tile grid exactly.
// ---------------------------------------------------------------------------

#define N_IN   128
#define NL     6
#define N_HID  320
#define DEG    32
#define N_OUT  18
#define BATCH  8192
#define N_TOT  (N_IN + NL * N_HID)   // 2048

#define BT      32                    // batch tile per workgroup
#define STRIDE  (N_TOT + 8)           // padded LDS row (halfs): 2056 -> 4-bank skew/row
#define THREADS 320                   // 10 waves (wave32): one 32x32 super-tile each

typedef __attribute__((ext_vector_type(16))) __bf16 v16bf;
typedef __attribute__((ext_vector_type(8)))  float  v8f;

__device__ __forceinline__ unsigned short f32_to_bf16(float f) {
  union { float f; unsigned u; } v; v.f = f;
  unsigned r = v.u + 0x7FFFu + ((v.u >> 16) & 1u);   // round-to-nearest-even
  return (unsigned short)(r >> 16);
}
__device__ __forceinline__ float bf16_to_f32(unsigned short h) {
  union { unsigned u; float f; } v; v.u = ((unsigned)h) << 16;
  return v.f;
}

// tanh(x) = sign(x) * (1 - 2/(exp(2|x|)+1)) using hw v_exp_f32 / v_rcp_f32
// (exact identity; avoids the branchy OCML tanhf expansion).
__device__ __forceinline__ float fast_tanh(float x) {
  float ax = fabsf(x);
  float e  = __builtin_amdgcn_exp2f(ax * 2.8853900817779268f); // 2*log2(e)
  float r  = 1.0f - 2.0f * __builtin_amdgcn_rcpf(e + 1.0f);
  return copysignf(r, x);
}

// ---------------------------------------------------------------------------
// Prep kernel: scatter sparse (src, w) into dense bf16 weight matrices.
// Layer l is K_l x 320, stored column-major (per-neuron column contiguous in
// K) so WMMA B-fragment loads are two contiguous 16B global loads per lane.
// One thread owns one (layer, neuron) column -> sequential scatter-add, no
// atomics, duplicate source indices accumulate exactly like the einsum.
// ---------------------------------------------------------------------------
__global__ void build_dense_w_kernel(const int* __restrict__ hid_src,
                                     const float* __restrict__ hid_w,
                                     unsigned short* __restrict__ Wd) {
  int t = blockIdx.x * blockDim.x + threadIdx.x;
  if (t >= NL * N_HID) return;
  int l = t / N_HID, n = t % N_HID;
  int K = N_IN + l * N_HID;

  long base = 0;
  for (int i = 0; i < l; ++i) base += (long)N_HID * (N_IN + i * N_HID);

  unsigned short* col = Wd + base + (long)n * K;
  for (int k = 0; k < K; ++k) col[k] = 0;            // bf16 zero

  const int*   src = hid_src + (long)(l * N_HID + n) * DEG;
  const float* w   = hid_w   + (long)(l * N_HID + n) * DEG;
  for (int d = 0; d < DEG; ++d) {
    int j = src[d];                                  // guaranteed j < K
    float cur = bf16_to_f32(col[j]);
    col[j] = f32_to_bf16(cur + w[d]);
  }
}

// ---------------------------------------------------------------------------
// Main kernel: one WG = 32 batches, full network depth in LDS.
// ---------------------------------------------------------------------------
__global__ __launch_bounds__(THREADS)
void policy_dense_wmma_kernel(const float* __restrict__ obs,
                              const float* __restrict__ hid_b,
                              const int*   __restrict__ out_src,
                              const float* __restrict__ out_w,
                              const float* __restrict__ out_b,
                              const unsigned short* __restrict__ Wd,
                              float* __restrict__ out) {
  extern __shared__ unsigned short acts[];           // [BT][STRIDE] bf16
  const int b0  = blockIdx.x * BT;
  const int tid = threadIdx.x;

  // ---- stage obs tile into LDS as bf16 ----
  for (int idx = tid; idx < BT * N_IN; idx += THREADS) {
    int b = idx / N_IN, j = idx % N_IN;
    acts[b * STRIDE + j] = f32_to_bf16(obs[(long)(b0 + b) * N_IN + j]);
  }
  __syncthreads();

  const int wave  = tid >> 5;                        // 0..9 -> neuron-tile pair
  const int lane  = tid & 31;
  const int lhalf = lane >> 4;                       // 0: lanes 0-15, 1: lanes 16-31
  const int lmod  = lane & 15;

  // This wave's two output columns (N of the two WMMA tiles) and two A rows.
  const int n0    = (wave * 2) * 16 + lmod;          // nt = 2*wave
  const int n1    = n0 + 16;                         // nt = 2*wave + 1
  const int arow0 = lmod * STRIDE;                   // bt = 0, M = lane%16
  const int arow1 = (16 + lmod) * STRIDE;            // bt = 1

  long wbase = 0;
  for (int l = 0; l < NL; ++l) {
    const int K   = N_IN + l * N_HID;                // GEMM depth for this layer
    const int off = K;                               // where this layer's outputs land

    const float bias0 = hid_b[l * N_HID + n0];       // C[M][N] init = b[N]
    const float bias1 = hid_b[l * N_HID + n1];
    v8f acc00 = { bias0, bias0, bias0, bias0, bias0, bias0, bias0, bias0 };
    v8f acc01 = { bias1, bias1, bias1, bias1, bias1, bias1, bias1, bias1 };
    v8f acc10 = acc00;
    v8f acc11 = acc01;

    const unsigned short* wcol0 = Wd + wbase + (long)n0 * K;
    const unsigned short* wcol1 = Wd + wbase + (long)n1 * K;

    for (int k0 = 0; k0 < K; k0 += 32) {
      // A (16x32 bf16), ISA layout: lanes<16 hold K{0..7,16..23},
      // lanes>=16 hold K{8..15,24..31} of row M=lane%16.
      union { v16bf v; uint4 q[2]; } a0, a1, bfr0, bfr1;
      const uint4* pa0 = (const uint4*)(acts + arow0 + k0 + lhalf * 8);
      const uint4* pa1 = (const uint4*)(acts + arow1 + k0 + lhalf * 8);
      a0.q[0] = pa0[0];  a0.q[1] = pa0[2];           // +0/+16 halfs
      a1.q[0] = pa1[0];  a1.q[1] = pa1[2];

      // B (32x16 bf16), column N=lane%16 per lane; lanes<16 K{0..7,16..23},
      // lanes>=16 K{8..15,24..31} -> two 16B loads of the dense column.
      const uint4* pb0 = (const uint4*)(wcol0 + k0 + lhalf * 8);
      const uint4* pb1 = (const uint4*)(wcol1 + k0 + lhalf * 8);
      bfr0.q[0] = pb0[0];  bfr0.q[1] = pb0[2];
      bfr1.q[0] = pb1[0];  bfr1.q[1] = pb1[2];

      // 2x2 register block: each fragment feeds two WMMAs.
      acc00 = __builtin_amdgcn_wmma_f32_16x16x32_bf16(
          false, a0.v, false, bfr0.v, (short)0, acc00, false, false);
      acc01 = __builtin_amdgcn_wmma_f32_16x16x32_bf16(
          false, a0.v, false, bfr1.v, (short)0, acc01, false, false);
      acc10 = __builtin_amdgcn_wmma_f32_16x16x32_bf16(
          false, a1.v, false, bfr0.v, (short)0, acc10, false, false);
      acc11 = __builtin_amdgcn_wmma_f32_16x16x32_bf16(
          false, a1.v, false, bfr1.v, (short)0, acc11, false, false);
    }

    // D layout: VGPR v holds M=v (lanes 0-15) / M=v+8 (lanes 16-31), N=lane%16.
    // Writes target columns >= K while reads were < K, so no intra-layer race.
    #pragma unroll
    for (int v = 0; v < 8; ++v) {
      const int r0 = (v + 8 * lhalf) * STRIDE + off;        // bt = 0 row
      const int r1 = (16 + v + 8 * lhalf) * STRIDE + off;   // bt = 1 row
      acts[r0 + n0] = f32_to_bf16(fast_tanh(acc00[v]));
      acts[r0 + n1] = f32_to_bf16(fast_tanh(acc01[v]));
      acts[r1 + n0] = f32_to_bf16(fast_tanh(acc10[v]));
      acts[r1 + n1] = f32_to_bf16(fast_tanh(acc11[v]));
    }
    __syncthreads();                                 // layer l+1 reads layer l outputs
    wbase += (long)N_HID * K;
  }

  // ---- output head: 18 neurons, 32-wide gather-dot (tiny -> VALU) ----
  for (int idx = tid; idx < BT * N_OUT; idx += THREADS) {
    int b = idx % BT, o = idx / BT;
    const int*   src = out_src + o * DEG;
    const float* w   = out_w  + o * DEG;
    float s = out_b[o];
    #pragma unroll 8
    for (int d = 0; d < DEG; ++d)
      s += bf16_to_f32(acts[b * STRIDE + src[d]]) * w[d];
    out[(long)(b0 + b) * N_OUT + o] = fast_tanh(s);
  }
}

// ---------------------------------------------------------------------------
extern "C" void kernel_launch(void* const* d_in, const int* in_sizes, int n_in,
                              void* d_out, int out_size, void* d_ws, size_t ws_size,
                              hipStream_t stream) {
  const float* obs     = (const float*)d_in[0];
  const int*   hid_src = (const int*)  d_in[1];
  const float* hid_w   = (const float*)d_in[2];
  const float* hid_b   = (const float*)d_in[3];
  const int*   out_src = (const int*)  d_in[4];
  const float* out_w   = (const float*)d_in[5];
  const float* out_b   = (const float*)d_in[6];
  unsigned short* Wd   = (unsigned short*)d_ws;      // ~3.56 MB bf16 dense weights

  build_dense_w_kernel<<<(NL * N_HID + 255) / 256, 256, 0, stream>>>(
      hid_src, hid_w, Wd);

  size_t lds_bytes = (size_t)BT * STRIDE * sizeof(unsigned short); // ~128.5 KB
  policy_dense_wmma_kernel<<<BATCH / BT, THREADS, lds_bytes, stream>>>(
      obs, hid_b, out_src, out_w, out_b, Wd, (float*)d_out);
}
